// LFPNetMC_34136400068625
// MI455X (gfx1250) — matrix operand order repaired and checked by reading, not compiled, not run on hardware
//
#include <hip/hip_runtime.h>

// ---------------------------------------------------------------------------
// Problem dims (compile-time; match reference)
// ---------------------------------------------------------------------------
#define B_ 4096
#define L_ 4096
#define H_ 2048
#define OUT_ 1024

typedef __bf16 bf16_t;
typedef __attribute__((ext_vector_type(16))) __bf16 v16bf;
typedef __attribute__((ext_vector_type(8)))  __bf16 v8bf;
typedef __attribute__((ext_vector_type(8)))  float  v8f;

// ---------------------------------------------------------------------------
// Split f32 -> bf16 hi/lo (Ootomo-style precision recovery for WMMA)
// ---------------------------------------------------------------------------
__global__ void lfp_split_bf16(const float* __restrict__ src,
                               bf16_t* __restrict__ hi, bf16_t* __restrict__ lo,
                               int n) {
    int i = blockIdx.x * blockDim.x + threadIdx.x;
    if (i < n) {
        float x = src[i];
        bf16_t h = (bf16_t)x;
        bf16_t l = (bf16_t)(x - (float)h);
        hi[i] = h;
        lo[i] = l;
    }
}

// ---------------------------------------------------------------------------
// Fused conv pipeline: (conv5+relu -> conv3+relu +res) x2 -> 1x1 conv+relu,
// emitting split-bf16 A-matrix for GEMM1. One block = one 256-wide row chunk.
// ---------------------------------------------------------------------------
#define CHUNK 256
#define HALO 6

__global__ void lfp_conv_fuse(const float* __restrict__ x,
                              const float* __restrict__ w5, const float* __restrict__ b5,
                              const float* __restrict__ w3, const float* __restrict__ b3,
                              const float* __restrict__ wk, const float* __restrict__ bk,
                              bf16_t* __restrict__ zhi, bf16_t* __restrict__ zlo) {
    __shared__ float s0[CHUNK + 2 * HALO]; // x, later t2
    __shared__ float s1[CHUNK + 2 * HALO]; // t1
    __shared__ float s2[CHUNK + 2 * HALO]; // y1
    const int b   = blockIdx.y;
    const int l0  = blockIdx.x * CHUNK;
    const int tid = threadIdx.x;

    float W5[5], W3[3];
#pragma unroll
    for (int j = 0; j < 5; ++j) W5[j] = w5[j];
#pragma unroll
    for (int j = 0; j < 3; ++j) W3[j] = w3[j];
    const float B5 = b5[0], B3 = b3[0], WK = wk[0], BK = bk[0];

    const float* xr = x + (size_t)b * L_;

    for (int i = tid; i < CHUNK + 2 * HALO; i += CHUNK) {
        int l = l0 + i - HALO;
        s0[i] = (l >= 0 && l < L_) ? xr[l] : 0.f;
    }
    __syncthreads();

    // t1 = relu(conv5(x)), valid on i in [2, CHUNK+10)
    for (int i = tid + 2; i < CHUNK + 10; i += CHUNK) {
        int l = l0 + i - HALO;
        float a = B5;
#pragma unroll
        for (int j = 0; j < 5; ++j) a += W5[j] * s0[i - 2 + j];
        a = a > 0.f ? a : 0.f;
        s1[i] = (l >= 0 && l < L_) ? a : 0.f;
    }
    __syncthreads();

    // y1 = relu(conv3(t1)) + x, valid on i in [3, CHUNK+9)
    for (int i = tid + 3; i < CHUNK + 9; i += CHUNK) {
        int l = l0 + i - HALO;
        float a = B3;
#pragma unroll
        for (int j = 0; j < 3; ++j) a += W3[j] * s1[i - 1 + j];
        a = (a > 0.f ? a : 0.f) + s0[i];
        s2[i] = (l >= 0 && l < L_) ? a : 0.f;
    }
    __syncthreads();

    // t2 = relu(conv5(y1)), valid on i in [5, CHUNK+7); reuse s0
    for (int i = tid + 5; i < CHUNK + 7; i += CHUNK) {
        int l = l0 + i - HALO;
        float a = B5;
#pragma unroll
        for (int j = 0; j < 5; ++j) a += W5[j] * s2[i - 2 + j];
        a = a > 0.f ? a : 0.f;
        s0[i] = (l >= 0 && l < L_) ? a : 0.f;
    }
    __syncthreads();

    // y2 = relu(conv3(t2)) + y1 ; z = relu(wk*y2 + bk); split-bf16 store
    {
        const int i = tid + HALO;
        float a = B3;
#pragma unroll
        for (int j = 0; j < 3; ++j) a += W3[j] * s0[i - 1 + j];
        a = (a > 0.f ? a : 0.f) + s2[i];
        float z = WK * a + BK;
        z = z > 0.f ? z : 0.f;
        bf16_t h = (bf16_t)z;
        bf16_t l = (bf16_t)(z - (float)h);
        size_t idx = (size_t)b * L_ + (size_t)(l0 + tid);
        zhi[idx] = h;
        zlo[idx] = l;
    }
}

// ---------------------------------------------------------------------------
// Fragment load: uniform SGPR base + 32-bit per-lane byte offset (saddr form);
// compile-time deltas fold into the 24-bit instruction offset.
// ---------------------------------------------------------------------------
__device__ __forceinline__ v8bf lfp_ld8(const bf16_t* __restrict__ base, uint32_t boff) {
    return *(const v8bf*)((const char*)base + boff);
}

__device__ __forceinline__ v16bf lfp_frag(const bf16_t* __restrict__ base,
                                          uint32_t boff, uint32_t d2) {
    union { v16bf v; v8bf h[2]; } u;
    u.h[0] = lfp_ld8(base, boff);
    u.h[1] = lfp_ld8(base, boff + d2);
    return u.v;
}

// ---------------------------------------------------------------------------
// bf16x3 split GEMM:  C[M,N] = A[M,K] * B[N,K]^T  (+bias, optional relu+split)
// Wave tile 32(M) x 64(N): 8 accumulators, 24 WMMAs / 24 b128 loads per K=32.
// Block = 128 threads = 4 waves stacked in M -> 128x64 block tile.
// A 16x32 frag: lane<16 -> K base 0, lane>=16 -> +8; halves at +0B and +32B.
// B 32x16 frag (row-major NxK source): lane<16 -> K 0, lane>=16 -> +8;
// halves at +0B and +32B.
// ---------------------------------------------------------------------------
template <bool RELU_SPLIT, int NN, int KK>
__global__ void lfp_gemm_bf16x3(const bf16_t* __restrict__ Ahi, const bf16_t* __restrict__ Alo,
                                const bf16_t* __restrict__ Bhi, const bf16_t* __restrict__ Blo,
                                const float* __restrict__ bias,
                                float* __restrict__ Cout,
                                bf16_t* __restrict__ Chi, bf16_t* __restrict__ Clo) {
    const int lane = threadIdx.x & 31;
    const int wave = threadIdx.x >> 5;
    const int m0 = blockIdx.y * 128 + wave * 32;
    const int n0 = blockIdx.x * 64;

    constexpr uint32_t AROW = 16u * KK * 2u;  // bytes: fragment row-block stride (m0 -> m0+16)
    constexpr uint32_t BCOL = 16u * KK * 2u;  // bytes: N-subtile stride

    v8f acc[2][4];
#pragma unroll
    for (int mi = 0; mi < 2; ++mi)
#pragma unroll
        for (int j = 0; j < 4; ++j)
            acc[mi][j] = (v8f){0.f, 0.f, 0.f, 0.f, 0.f, 0.f, 0.f, 0.f};

    uint32_t aoff = ((uint32_t)(m0 + (lane & 15)) * KK + (uint32_t)((lane >> 4) << 3)) * 2u;
    uint32_t boff = ((uint32_t)(n0 + (lane & 15)) * KK + (uint32_t)((lane >> 4) << 3)) * 2u;

    for (int k0 = 0; k0 < KK; k0 += 32) {
        v16bf ah0 = lfp_frag(Ahi, aoff, 32u);
        v16bf al0 = lfp_frag(Alo, aoff, 32u);
        v16bf ah1 = lfp_frag(Ahi, aoff + AROW, 32u);
        v16bf al1 = lfp_frag(Alo, aoff + AROW, 32u);
#pragma unroll
        for (int j = 0; j < 4; ++j) {
            v16bf bh = lfp_frag(Bhi, boff + (uint32_t)j * BCOL, 32u);
            v16bf bl = lfp_frag(Blo, boff + (uint32_t)j * BCOL, 32u);
            // interleave accumulators: no back-to-back WMMA on the same D regs
            acc[0][j] = __builtin_amdgcn_wmma_f32_16x16x32_bf16(false, ah0, false, bh,
                                                                (short)0, acc[0][j], false, false);
            acc[1][j] = __builtin_amdgcn_wmma_f32_16x16x32_bf16(false, ah1, false, bh,
                                                                (short)0, acc[1][j], false, false);
            acc[0][j] = __builtin_amdgcn_wmma_f32_16x16x32_bf16(false, ah0, false, bl,
                                                                (short)0, acc[0][j], false, false);
            acc[1][j] = __builtin_amdgcn_wmma_f32_16x16x32_bf16(false, ah1, false, bl,
                                                                (short)0, acc[1][j], false, false);
            acc[0][j] = __builtin_amdgcn_wmma_f32_16x16x32_bf16(false, al0, false, bh,
                                                                (short)0, acc[0][j], false, false);
            acc[1][j] = __builtin_amdgcn_wmma_f32_16x16x32_bf16(false, al1, false, bh,
                                                                (short)0, acc[1][j], false, false);
        }
        aoff += 64u;
        boff += 64u;
    }

    // Epilogue. C/D layout: VGPR r -> row m + r + (lane>=16 ? 8 : 0), col = n0+16j+(lane&15)
    const int colb   = lane & 15;
    const int rowoff = (lane >> 4) ? 8 : 0;
#pragma unroll
    for (int mi = 0; mi < 2; ++mi) {
#pragma unroll
        for (int j = 0; j < 4; ++j) {
            const int col = n0 + 16 * j + colb;
            const float bv = bias[col];
#pragma unroll
            for (int r = 0; r < 8; ++r) {
                const int row = m0 + mi * 16 + rowoff + r;
                float v = acc[mi][j][r] + bv;
                const size_t idx = (size_t)row * NN + col;
                if (RELU_SPLIT) {
                    v = v > 0.f ? v : 0.f;
                    bf16_t h = (bf16_t)v;
                    bf16_t l = (bf16_t)(v - (float)h);
                    Chi[idx] = h;
                    Clo[idx] = l;
                } else {
                    Cout[idx] = v;
                }
            }
        }
    }
}

// ---------------------------------------------------------------------------
// Launch
// ---------------------------------------------------------------------------
extern "C" void kernel_launch(void* const* d_in, const int* in_sizes, int n_in,
                              void* d_out, int out_size, void* d_ws, size_t ws_size,
                              hipStream_t stream) {
    const float* x     = (const float*)d_in[0];
    const float* cb_w1 = (const float*)d_in[1];
    const float* cb_b1 = (const float*)d_in[2];
    const float* cb_w2 = (const float*)d_in[3];
    const float* cb_b2 = (const float*)d_in[4];
    const float* ck1_w = (const float*)d_in[5];
    const float* ck1_b = (const float*)d_in[6];
    const float* fc1_w = (const float*)d_in[7];
    const float* fc1_b = (const float*)d_in[8];
    const float* fc2_w = (const float*)d_in[9];
    const float* fc2_b = (const float*)d_in[10];

    char* ws = (char*)d_ws;
    bf16_t* zhi  = (bf16_t*)ws; ws += (size_t)B_ * L_ * sizeof(bf16_t);
    bf16_t* zlo  = (bf16_t*)ws; ws += (size_t)B_ * L_ * sizeof(bf16_t);
    bf16_t* w1hi = (bf16_t*)ws; ws += (size_t)H_ * L_ * sizeof(bf16_t);
    bf16_t* w1lo = (bf16_t*)ws; ws += (size_t)H_ * L_ * sizeof(bf16_t);
    bf16_t* hhi  = (bf16_t*)ws; ws += (size_t)B_ * H_ * sizeof(bf16_t);
    bf16_t* hlo  = (bf16_t*)ws; ws += (size_t)B_ * H_ * sizeof(bf16_t);
    bf16_t* w2hi = (bf16_t*)ws; ws += (size_t)OUT_ * H_ * sizeof(bf16_t);
    bf16_t* w2lo = (bf16_t*)ws; ws += (size_t)OUT_ * H_ * sizeof(bf16_t);

    // 1) weight conversion to split-bf16
    {
        int n1 = H_ * L_;
        lfp_split_bf16<<<(n1 + 255) / 256, 256, 0, stream>>>(fc1_w, w1hi, w1lo, n1);
        int n2 = OUT_ * H_;
        lfp_split_bf16<<<(n2 + 255) / 256, 256, 0, stream>>>(fc2_w, w2hi, w2lo, n2);
    }

    // 2) fused conv pipeline -> split-bf16 activations
    lfp_conv_fuse<<<dim3(L_ / CHUNK, B_), CHUNK, 0, stream>>>(
        x, cb_w1, cb_b1, cb_w2, cb_b2, ck1_w, ck1_b, zhi, zlo);

    // 3) fc1 (+bias, relu) -> split-bf16 hidden
    lfp_gemm_bf16x3<true, H_, L_><<<dim3(H_ / 64, B_ / 128), 128, 0, stream>>>(
        zhi, zlo, w1hi, w1lo, fc1_b, nullptr, hhi, hlo);

    // 4) fc2 (+bias) -> f32 output
    lfp_gemm_bf16x3<false, OUT_, H_><<<dim3(OUT_ / 64, B_ / 128), 128, 0, stream>>>(
        hhi, hlo, w2hi, w2lo, fc2_b, (float*)d_out, nullptr, nullptr);
}
